// DSVF_84335977824459
// MI455X (gfx1250) — matrix operand order repaired and checked by reading, not compiled, not run on hardware
//
#include <hip/hip_runtime.h>

// ---------------------------------------------------------------------------
// DSVF via FFT-overlap-add  ==  GEMM against a circulant slice of the aliased
// biquad impulse response:
//   out(8192 x 2048) = U(8192 x 4096) * W(4096 x 2048)
//   U row r (= batch bi, seg si): contiguous window x[bi][(si-1)*2048 .. +4096)
//   W[k][n] = h[(n - k + 2048) mod 4096],  h = time-aliased biquad IR
//
// MI455X plan: compute-bound (137 GFLOP fp32-equivalent); bf16 split-precision
// (a = a_hi + a_lo) on V_WMMA_F32_16X16X32_BF16: hi*hi + hi*lo + lo*hi gives
// ~2^-16 relative accuracy at 3/8 the cost of the native f32 16x16x4 WMMA.
// All operand prep (hi/lo splits, W swizzle, zero padding for the first
// segment's window) is hoisted into memory-bound pre-kernels so the GEMM
// inner loop is branch-free {24x b128 loads + 24x v_wmma} per K-step.
// Working set (xhi/xlo 68MB + W 34MB + out 67MB) fits the 192MB L2.
//
// Workspace layout (needs ~102 MB):
//   [0)       h    : float[4096]         aliased impulse response
//   [16 KB)   Wfrag_hi: __bf16[8388608]  W pre-swizzled into B-fragment order
//   [+16 MB)  Wfrag_lo: __bf16[8388608]    layout [kt][ntile][lane][e]
//   [+16 MB)  xpad_hi: __bf16[64*264192] per-batch: [2048 zeros][x row, bf16]
//   [+33 MB)  xpad_lo: __bf16[64*264192]
// ---------------------------------------------------------------------------

typedef __attribute__((ext_vector_type(16))) __bf16 v16bf;
typedef __attribute__((ext_vector_type(8)))  __bf16 v8bf;
typedef __attribute__((ext_vector_type(8)))  float  v8f;

#define NSEG    128
#define SEGLEN  2048
#define NFFT    4096
#define MROWS   8192              // 64 batches * 128 segments
#define KTOT    4096
#define NTOT    2048
#define NBATCH  64
#define W_ELEMS (8388608ull)      // 4096*2048
#define PADSTRIDE (SEGLEN + NSEG * SEGLEN)   // 264192 elems per padded batch
#define XP_ELEMS ((unsigned long long)NBATCH * PADSTRIDE)  // 16,908,288

// ---------------------------------------------------------------------------
// Kernel 1: biquad coefficients + aliased impulse response h[4096].
// h_alias[n] = sum_m h_inf[n + 4096 m]; poles are strictly inside |z|=1
// (Rt = softplus(R) > 0), so 4 periods of the recurrence is exhaustive.
// ---------------------------------------------------------------------------
__global__ void dsvf_impulse_kernel(const float* g, const float* R,
                                    const float* m_hp, const float* m_bp,
                                    const float* m_lp, float* __restrict__ h) {
    const int t = threadIdx.x;
    for (int i = t; i < NFFT; i += blockDim.x) h[i] = 0.0f;
    __syncthreads();
    if (t == 0) {
        const float PI_F = 3.14159265358979323846f;
        float gv  = g[0];
        float sig = 1.0f / (1.0f + expf(-gv));
        float gt  = tanf(PI_F * sig * 0.5f);
        float Rt  = log1pf(expf(R[0]));          // softplus
        float g2  = gt * gt;
        float mh = m_hp[0], mb = m_bp[0], ml = m_lp[0];
        float b0 = g2 * ml + gt * mb + mh;
        float b1 = 2.0f * g2 * ml - 2.0f * mh;
        float b2 = g2 * ml - gt * mb + mh;
        float a0 = g2 + 2.0f * Rt * gt + 1.0f;
        float a1 = 2.0f * g2 - 2.0f;
        float a2 = g2 - 2.0f * Rt * gt + 1.0f;
        float inv = 1.0f / a0;
        float y1 = 0.0f, y2 = 0.0f;
        for (int n = 0; n < 4 * NFFT; ++n) {
            float xb = (n == 0) ? b0 : (n == 1) ? b1 : (n == 2) ? b2 : 0.0f;
            float y  = (xb - a1 * y1 - a2 * y2) * inv;
            h[n & (NFFT - 1)] += y;              // fold == time aliasing
            y2 = y1; y1 = y;
        }
    }
}

// ---------------------------------------------------------------------------
// Kernel 2: materialize W in WMMA B-operand fragment order, split bf16 hi/lo.
// Fragment (kt, nt): element e of lane L holds
//   k = kt*32 + (e&7) + 16*(e>=8) + 8*(L>=16),  n = nt*16 + (L&15)   (32x16 bf16 B layout)
// Flat: (((kt*128 + nt)*32) + L)*16 + e  ->  each lane's 16 elements are 32
// contiguous bytes => 2x b128 coalesced loads per fragment in the GEMM.
// ---------------------------------------------------------------------------
__global__ void dsvf_wswizzle_kernel(const float* __restrict__ h,
                                     __bf16* __restrict__ whi,
                                     __bf16* __restrict__ wlo) {
    const long t = (long)blockIdx.x * blockDim.x + threadIdx.x; // < 8388608
    const int e  = (int)t & 15;
    const int L  = ((int)t >> 4) & 31;
    const int nt = ((int)t >> 9) & 127;
    const int kt = (int)(t >> 16);
    const int k  = kt * 32 + (e & 7) + ((e >= 8) ? 16 : 0) + ((L >= 16) ? 8 : 0);
    const int n  = nt * 16 + (L & 15);
    const float f = h[(n - k + 2048) & (NFFT - 1)];
    const __bf16 hi = (__bf16)f;
    whi[t] = hi;
    wlo[t] = (__bf16)(f - (float)hi);
}

// ---------------------------------------------------------------------------
// Kernel 3: bf16 hi/lo split of x with a 2048-elem zero prefix per batch, so
// the segment-0 window (which reaches "before" the batch) needs no predicate
// in the GEMM.  Memory-bound (~135 MB total traffic, ~6 us at 23.3 TB/s).
// Grid: (PADSTRIDE/8/256, NBATCH) -> blockIdx.y == batch, no divisions.
// ---------------------------------------------------------------------------
__global__ void dsvf_xsplit_kernel(const float* __restrict__ x,
                                   __bf16* __restrict__ xhi,
                                   __bf16* __restrict__ xlo) {
    const size_t bi = blockIdx.y;
    const size_t p  = ((size_t)blockIdx.x * blockDim.x + threadIdx.x) * 8;
    v8bf hi, lo;
    if (p < SEGLEN) {                          // zero prefix (pad region)
#pragma unroll
        for (int e = 0; e < 8; ++e) { hi[e] = (__bf16)0.0f; lo[e] = (__bf16)0.0f; }
    } else {
        const float* src = x + bi * (NSEG * SEGLEN) + (p - SEGLEN);
        const float4 a = *(const float4*)(src);
        const float4 b = *(const float4*)(src + 4);
        const float va[8] = {a.x, a.y, a.z, a.w, b.x, b.y, b.z, b.w};
#pragma unroll
        for (int e = 0; e < 8; ++e) {
            __bf16 hv = (__bf16)va[e];
            hi[e] = hv;
            lo[e] = (__bf16)(va[e] - (float)hv);
        }
    }
    *(v8bf*)(xhi + bi * PADSTRIDE + p) = hi;
    *(v8bf*)(xlo + bi * PADSTRIDE + p) = lo;
}

// ---------------------------------------------------------------------------
// Kernel 4: GEMM.  256 threads = 8 waves arranged 4(M) x 2(N); block tile
// 128(M) x 128(N); each wave owns 32 rows (2 M-tiles) x 64 cols (4 N-tiles)
// = 8 accumulators -> 24 WMMA per 24 b128 loads per K-step, branch-free.
// Grid: (2048/128, 8192/128) = (16, 64); blockIdx.y == batch, so all rows a
// wave touches share one batch and segment indexing is trivial.
// A fragments (16x32 bf16 layout): lane L holds row (L&15); K runs
// {k0..k0+7} and {k0+16..k0+23}, k0 = kc + 8*(L>=16) -> 2x 16B loads each,
// read from the zero-padded window xpad + bi*PADSTRIDE + si*2048.
// ---------------------------------------------------------------------------
__global__ __launch_bounds__(256)
void dsvf_gemm_kernel(const __bf16* __restrict__ xhi,
                      const __bf16* __restrict__ xlo,
                      const __bf16* __restrict__ whi,
                      const __bf16* __restrict__ wlo,
                      float* __restrict__ out) {
    const int lane = threadIdx.x & 31;
    const int wave = threadIdx.x >> 5;
    const int wq   = wave & 3;         // M quadrant (32 rows each)
    const int wn   = wave >> 2;        // N half (64 cols each)
    const int half = lane >> 4;        // 16-lane group
    const int lrow = lane & 15;
    const int bx   = blockIdx.x;       // N block (128 cols)
    const int by   = blockIdx.y;       // batch  (128 rows)

    // Window base per M-tile: zero-padded, valid for si==0 too.
    size_t uoff[2];
#pragma unroll
    for (int mt = 0; mt < 2; ++mt) {
        const int si = wq * 32 + mt * 16 + lrow;
        uoff[mt] = (size_t)by * PADSTRIDE + (size_t)si * SEGLEN;
    }
    const int ntbase = bx * 8 + wn * 4;   // first of this wave's 4 N-tiles

    v8f c[2][4] = {};                     // 8 independent 16x16 accumulators

    for (int kt = 0; kt < KTOT / 32; ++kt) {
        const int k0 = kt * 32 + half * 8;

        // ---- A fragments: 2 M-tiles, hi and lo, 2x 16B loads each ----
        v16bf ahi[2], alo[2];
#pragma unroll
        for (int mt = 0; mt < 2; ++mt) {
            const v8bf h1 = *(const v8bf*)(xhi + uoff[mt] + k0);
            const v8bf h2 = *(const v8bf*)(xhi + uoff[mt] + k0 + 16);
            const v8bf l1 = *(const v8bf*)(xlo + uoff[mt] + k0);
            const v8bf l2 = *(const v8bf*)(xlo + uoff[mt] + k0 + 16);
#pragma unroll
            for (int e = 0; e < 8; ++e) {
                ahi[mt][e] = h1[e]; ahi[mt][e + 8] = h2[e];
                alo[mt][e] = l1[e]; alo[mt][e + 8] = l2[e];
            }
        }

        // ---- B fragments: pre-swizzled, 2x b128 per lane each ----
        v16bf bh[4], bl[4];
#pragma unroll
        for (int nt = 0; nt < 4; ++nt) {
            const size_t fo =
                (((size_t)kt * 128 + (size_t)(ntbase + nt)) * 32 + lane) * 16;
            bh[nt] = *reinterpret_cast<const v16bf*>(whi + fo);
            bl[nt] = *reinterpret_cast<const v16bf*>(wlo + fo);
        }
        if (kt + 1 < KTOT / 32) {          // pull next K-slice toward the WGP
            const size_t fn =
                (((size_t)(kt + 1) * 128 + (size_t)ntbase) * 32 + lane) * 16;
            __builtin_prefetch(whi + fn, 0, 1);
            __builtin_prefetch(wlo + fn, 0, 1);
        }

        // ---- split-precision accumulate: hi*hi + hi*lo + lo*hi.
        // 8 distinct accumulators per pass -> consecutive WMMAs never share
        // a D register (sidesteps the 5-NOP D->A/B RAW hazard chain). ----
#pragma unroll
        for (int mt = 0; mt < 2; ++mt)
#pragma unroll
            for (int nt = 0; nt < 4; ++nt)
                c[mt][nt] = __builtin_amdgcn_wmma_f32_16x16x32_bf16(
                    false, ahi[mt], false, bh[nt], (short)0, c[mt][nt],
                    false, false);
#pragma unroll
        for (int mt = 0; mt < 2; ++mt)
#pragma unroll
            for (int nt = 0; nt < 4; ++nt)
                c[mt][nt] = __builtin_amdgcn_wmma_f32_16x16x32_bf16(
                    false, ahi[mt], false, bl[nt], (short)0, c[mt][nt],
                    false, false);
#pragma unroll
        for (int mt = 0; mt < 2; ++mt)
#pragma unroll
            for (int nt = 0; nt < 4; ++nt)
                c[mt][nt] = __builtin_amdgcn_wmma_f32_16x16x32_bf16(
                    false, alo[mt], false, bh[nt], (short)0, c[mt][nt],
                    false, false);
    }

    // ---- store D tiles: f32 16x16 layout -> lane L, vgpr v holds
    //      (m = v + 8*(L>=16), n = L&15) ----
#pragma unroll
    for (int mt = 0; mt < 2; ++mt) {
        const size_t rbase =
            (size_t)by * NSEG + (size_t)wq * 32 + (size_t)mt * 16 + 8 * half;
#pragma unroll
        for (int nt = 0; nt < 4; ++nt) {
            const size_t ncol = (size_t)(ntbase + nt) * 16 + lrow;
#pragma unroll
            for (int v = 0; v < 8; ++v)
                out[(rbase + v) * NTOT + ncol] = c[mt][nt][v];
        }
    }
}

// ---------------------------------------------------------------------------
extern "C" void kernel_launch(void* const* d_in, const int* in_sizes, int n_in,
                              void* d_out, int out_size, void* d_ws, size_t ws_size,
                              hipStream_t stream) {
    const float* x    = (const float*)d_in[0];
    const float* g    = (const float*)d_in[1];
    const float* R    = (const float*)d_in[2];
    const float* m_hp = (const float*)d_in[3];
    const float* m_bp = (const float*)d_in[4];
    const float* m_lp = (const float*)d_in[5];
    float* out = (float*)d_out;

    float*  h   = (float*)d_ws;
    __bf16* whi = (__bf16*)((char*)d_ws + 16384);
    __bf16* wlo = whi + W_ELEMS;
    __bf16* xhi = wlo + W_ELEMS;
    __bf16* xlo = xhi + XP_ELEMS;
    // requires ws_size >= 16 KB + 2*16 MiB + 2*32.25 MiB  (~102 MB)

    dsvf_impulse_kernel<<<1, 256, 0, stream>>>(g, R, m_hp, m_bp, m_lp, h);
    dsvf_wswizzle_kernel<<<(unsigned)(W_ELEMS / 256), 256, 0, stream>>>(h, whi, wlo);
    dsvf_xsplit_kernel<<<dim3(PADSTRIDE / (8 * 256), NBATCH), 256, 0, stream>>>(
        x, xhi, xlo);
    dsvf_gemm_kernel<<<dim3(NTOT / 128, MROWS / NSEG), 256, 0, stream>>>(
        xhi, xlo, whi, wlo, out);
}
